// GConv2d_51453708206689
// MI455X (gfx1250) — matrix operand order repaired and checked by reading, not compiled, not run on hardware
//
#include <hip/hip_runtime.h>

typedef _Float16 v8h  __attribute__((ext_vector_type(8)));
typedef _Float16 v16h __attribute__((ext_vector_type(16)));
typedef float    v4f  __attribute__((ext_vector_type(4)));
typedef float    v8f  __attribute__((ext_vector_type(8)));

#define OUT_CH 16
#define IN_CH  16
#define TRANS  4
#define KS     3
#define CIN    64    // IN_CH*TRANS
#define COUT   64    // OUT_CH*TRANS
#define KDIM   576   // CIN*KS*KS
#define BATCH  32
#define HDIM   128
#define WDIM   128
#define HO     126
#define WO     126
#define HTILE  4                 // output rows per workgroup
#define NROWS  (HTILE + KS - 1)  // staged input rows = 6

// ---------------------------------------------------------------------------
// Prepass: gather filters through the p4 group-element index maps into the
// f16 GEMM A matrix: ftA[M][k], M = o*4+sp (64 rows), k = (u*3+v)*64 + (i*4+s).
// K ordering gives each 32-wide WMMA K-step a fixed (kh,kw) and 32 consecutive
// input channels (contiguous operand fetches).
// ---------------------------------------------------------------------------
__global__ void gconv_prep_kernel(const float* __restrict__ filters,
                                  const int*   __restrict__ idx_s,
                                  const int*   __restrict__ idx_u,
                                  const int*   __restrict__ idx_v,
                                  _Float16*    __restrict__ ftA) {
    int gid = blockIdx.x * blockDim.x + threadIdx.x;
    if (gid >= COUT * KDIM) return;
    int M = gid / KDIM;
    int k = gid - M * KDIM;
    int khkw = k >> 6;           // (u*3+v) in 0..8
    int c    = k & 63;           // i*4+s
    int u = khkw / 3, v = khkw % 3;
    int i = c >> 2,  s = c & 3;
    int o = M >> 2,  sp = M & 3;
    int ii = ((sp * TRANS + s) * KS + u) * KS + v;   // idx_* are (St,S,K,K)
    int ss = idx_s[ii];
    int uu = idx_u[ii];
    int vv = idx_v[ii];
    float wv = filters[(((o * IN_CH + i) * TRANS + ss) * KS + uu) * KS + vv];
    ftA[(size_t)M * KDIM + k] = (_Float16)wv;
}

// ---------------------------------------------------------------------------
// Main implicit-GEMM conv kernel.
// Grid: (w_tile=2, h_tile=32, b=32). Block: 256 threads = 8 waves (wave32).
// Workgroup tile: 64 out-channels x 4 output rows x 64 w positions.
// Wave tiling: 4 (M) x 2 (N) waves; each wave holds 4x2 = 8 f32 16x16
// accumulators. K loop: 18 steps of 32; each step issues 8 WMMAs per wave
// (A fragment reused 8x).
// ---------------------------------------------------------------------------
__global__ __launch_bounds__(256)
void gconv_wmma_kernel(const float*    __restrict__ x,
                       const _Float16* __restrict__ ftA,
                       float*          __restrict__ out) {
    const int w0 = blockIdx.x * 64;
    const int h0 = blockIdx.y * HTILE;
    const int b  = blockIdx.z;

    // LDS: staged input transposed to [row][col][ic] (f16): a WMMA B fragment
    // per lane (16 consecutive ic at fixed column) is two ds_load_b128, and
    // the kw tap shift moves along the col index, keeping 32B alignment.
    __shared__ __align__(32) _Float16 xt[NROWS][72][CIN];   // 55,296 bytes

    const int tid = threadIdx.x;

    // ---- Stage + convert f32 -> f16 into LDS -------------------------------
    // 6 passes (one staged row each); 4 threads per ic-row; float4 loads
    // coalesced along w. No integer divisions anywhere.
    {
        const float* xb = x + (size_t)b * CIN * HDIM * WDIM;
        const int ic = tid >> 2;          // 0..63
        const int q  = tid & 3;           // vec-index subset
        #pragma unroll
        for (int p = 0; p < NROWS; ++p) {
            int hrow = h0 + p;
            if (hrow > HDIM - 1) hrow = HDIM - 1;        // clamp bottom edge
            const float* src = xb + ((size_t)ic * HDIM + hrow) * WDIM;
            for (int vi = q; vi < 17; vi += 4) {         // 68 cols as 17 vec4
                int wbase = w0 + vi * 4;
                if (wbase > WDIM - 4) wbase = WDIM - 4;  // clamp right halo
                v4f v = *(const v4f*)(src + wbase);
                const int c = vi * 4;
                xt[p][c + 0][ic] = (_Float16)v.x;
                xt[p][c + 1][ic] = (_Float16)v.y;
                xt[p][c + 2][ic] = (_Float16)v.z;
                xt[p][c + 3][ic] = (_Float16)v.w;
            }
        }
    }
    __syncthreads();

    const int lane = tid & 31;
    const int wave = tid >> 5;
    const int m0   = (wave & 3) * 16;                   // M tile base
    const int nb   = (wave >> 2) * 32;                  // N tile base
    const int lrow = lane & 15;
    const int hi   = (lane >> 4) & 1;                   // lane half

    v8f acc[HTILE][2] = {};

    // A fragment: lanes 0-15 take K {0..15},
    // lanes 16-31 take K {16..31}; two 16B global loads per K-step.
    const _Float16* arow = ftA + (size_t)(m0 + lrow) * KDIM + hi * 16;

    for (int t = 0; t < 18; ++t) {
        const int khkw = t >> 1;
        const int c0   = (t & 1) * 32;
        const int kh   = khkw / 3;
        const int kw   = khkw % 3;

        v8h a0 = *(const v8h*)(arow + t * 32);
        v8h a1 = *(const v8h*)(arow + t * 32 + 8);
        v16h a;
        #pragma unroll
        for (int i = 0; i < 8; ++i) { a[i] = a0[i]; a[8 + i] = a1[i]; }

        const int icb = c0 + hi * 16;                   // 32B-aligned in LDS
        #pragma unroll
        for (int dh = 0; dh < HTILE; ++dh) {
            const v16h bt0 = *(const v16h*)&xt[kh + dh][nb +      lrow + kw][icb];
            const v16h bt1 = *(const v16h*)&xt[kh + dh][nb + 16 + lrow + kw][icb];
            acc[dh][0] = __builtin_amdgcn_wmma_f32_16x16x32_f16(
                             false, a, false, bt0, (short)0, acc[dh][0], false, false);
            acc[dh][1] = __builtin_amdgcn_wmma_f32_16x16x32_f16(
                             false, a, false, bt1, (short)0, acc[dh][1], false, false);
        }
    }

    // ---- Store per documented C/D layout: VGPR r -> row m0+r+8*hi, lane%16 -> col
    #pragma unroll
    for (int dh = 0; dh < HTILE; ++dh) {
        const int hh = h0 + dh;
        if (hh >= HO) continue;                          // bottom-edge mask (uniform)
        #pragma unroll
        for (int r = 0; r < 8; ++r) {
            const int M = m0 + r + hi * 8;
            float* orow = out + ((size_t)(b * COUT + M) * HO + hh) * WO;
            const int wA = w0 + nb + lrow;
            if (wA < WO) orow[wA] = acc[dh][0][r];
            const int wB = wA + 16;
            if (wB < WO) orow[wB] = acc[dh][1][r];
        }
    }
}

// ---------------------------------------------------------------------------
extern "C" void kernel_launch(void* const* d_in, const int* in_sizes, int n_in,
                              void* d_out, int out_size, void* d_ws, size_t ws_size,
                              hipStream_t stream) {
    const float* x       = (const float*)d_in[0];
    const float* filters = (const float*)d_in[1];
    const int*   idx_s   = (const int*)d_in[2];
    const int*   idx_u   = (const int*)d_in[3];
    const int*   idx_v   = (const int*)d_in[4];
    float*       out     = (float*)d_out;
    _Float16*    ftA     = (_Float16*)d_ws;   // 64*576*2 = 73,728 bytes

    gconv_prep_kernel<<<(COUT * KDIM + 255) / 256, 256, 0, stream>>>(
        filters, idx_s, idx_u, idx_v, ftA);

    gconv_wmma_kernel<<<dim3(2, (HO + HTILE - 1) / HTILE, BATCH), 256, 0, stream>>>(
        x, ftA, out);
}
